// PaLM_403726926045
// MI455X (gfx1250) — compile-verified, matrix operand order checked by reading, not run on hardware
//
#include <hip/hip_runtime.h>

// Model dims (fixed by reference)
#define NSEQ   2048
#define DM     2048
#define HEADS  16
#define DHD    128
#define FFI    8192
#define VOC    32000
#define NDEPTH 4

#define GPAD   40     // padded row stride (ushorts) for 32-wide k rows (16B-aligned, bank-spread)
#define KPAD   136    // padded row stride for 128-wide d rows

typedef __attribute__((ext_vector_type(16))) __bf16 bf16x16;
typedef __attribute__((ext_vector_type(8)))  float  v8f;

union Frag {
    uint4   u[2];
    bf16x16 v;
};

__device__ __forceinline__ unsigned short f2bf(float f) {
    unsigned u = __float_as_uint(f);
    u += 0x7FFFu + ((u >> 16) & 1u);           // round-to-nearest-even
    return (unsigned short)(u >> 16);
}

__device__ __forceinline__ v8f v8f_zero() {
    v8f z = {0.f, 0.f, 0.f, 0.f, 0.f, 0.f, 0.f, 0.f};
    return z;
}

// Per the CDNA5 16-bit A/B fragment tables, a lane's 16 bf16 values of a 16x32
// subtile are two contiguous 8-element runs of row (lane&15):
//   k in [grp*8, grp*8+8) and [16+grp*8, 16+grp*8+8), grp = lane>>4.
// So a row-major [mn][k] LDS tile is read with two b128 loads at +0 / +16 elems.
__device__ __forceinline__ void load_frag(const unsigned short* rowbase, int lane, Frag& f) {
    const unsigned short* p = rowbase + ((lane >> 4) << 3);
    f.u[0] = *(const uint4*)p;
    f.u[1] = *(const uint4*)(p + 16);
}

// ---------------------------------------------------------------------------
// Generic tiled GEMM: C[M,N] = A[M,K](bf16) * B (f32 weights -> bf16)
//   BT=false : B is [K,N] row-major.   BT=true : B element (k,n) = Bp[n*K+k].
//   RESID    : add R[m,n] (f32).  OUTF32: write f32 C.  OUTBF: write bf16 C.
// Block = 128x128, 256 threads (8 waves), wave tile = 32x64 (2x4 WMMA tiles).
// Double-buffered LDS; next tile's global loads overlap current tile's WMMAs.
// ---------------------------------------------------------------------------
__device__ __forceinline__ void load_a_regs(const unsigned short* __restrict__ A,
                                            int K, int bm, int k0, int tid,
                                            uint4& r0, uint4& r1) {
    const unsigned short* src = A + (size_t)(bm + (tid >> 1)) * K + k0 + ((tid & 1) << 4);
    r0 = *(const uint4*)src;
    r1 = *(const uint4*)(src + 8);
}

template <bool BT>
__device__ __forceinline__ void load_b_regs(const float* __restrict__ B,
                                            int N, int K, int bn, int k0, int tid,
                                            float rb[16]) {
    if (BT) {
        const float* src = B + (size_t)(bn + (tid >> 1)) * K + k0 + ((tid & 1) << 4);
#pragma unroll
        for (int q = 0; q < 4; ++q) {
            float4 f = *(const float4*)(src + 4 * q);
            rb[4 * q + 0] = f.x; rb[4 * q + 1] = f.y;
            rb[4 * q + 2] = f.z; rb[4 * q + 3] = f.w;
        }
    } else {
        const float* src = B + (size_t)(k0 + ((tid >> 7) << 4)) * N + bn + (tid & 127);
#pragma unroll
        for (int e = 0; e < 16; ++e) rb[e] = src[(size_t)e * N];   // coalesced across lanes
    }
}

template <bool BT, bool RESID, bool OUTF32, bool OUTBF>
__global__ __launch_bounds__(256) void gemm_bf16_wmma(
    const unsigned short* __restrict__ A, const float* __restrict__ B,
    float* __restrict__ Cf, unsigned short* __restrict__ Cb,
    const float* __restrict__ R, int M, int N, int K)
{
    __shared__ unsigned short lA[2][128 * GPAD];
    __shared__ unsigned short lB[2][128 * GPAD];

    const int tid  = threadIdx.x;
    const int lane = tid & 31;
    const int wave = tid >> 5;
    const int bm   = blockIdx.x << 7;
    const int bn   = blockIdx.y << 7;
    const int wsm  = (wave >> 1) << 1;    // A subtile base (0,2,4,6)
    const int wsn  = (wave & 1) << 2;     // B subtile base (0,4)

    v8f acc[2][4];
#pragma unroll
    for (int i = 0; i < 2; ++i)
#pragma unroll
        for (int j = 0; j < 4; ++j) acc[i][j] = v8f_zero();

    uint4 ra0, ra1;
    float rb[16];
    load_a_regs(A, K, bm, 0, tid, ra0, ra1);
    load_b_regs<BT>(B, N, K, bn, 0, tid, rb);

    int cur = 0;
    for (int k0 = 0; k0 < K; k0 += 32) {
        // ---- commit staged tile to LDS buffer `cur` (row-major, padded) ----
        {
            unsigned short* da = &lA[cur][(tid >> 1) * GPAD + ((tid & 1) << 4)];
            *(uint4*)da = ra0;
            *(uint4*)(da + 8) = ra1;
        }
        {
            unsigned pk[8];
#pragma unroll
            for (int e = 0; e < 8; ++e)
                pk[e] = (unsigned)f2bf(rb[2 * e]) | ((unsigned)f2bf(rb[2 * e + 1]) << 16);
            uint4 b0, b1;
            b0.x = pk[0]; b0.y = pk[1]; b0.z = pk[2]; b0.w = pk[3];
            b1.x = pk[4]; b1.y = pk[5]; b1.z = pk[6]; b1.w = pk[7];
            unsigned short* db = BT ? &lB[cur][(tid >> 1) * GPAD + ((tid & 1) << 4)]
                                    : &lB[cur][(tid & 127) * GPAD + ((tid >> 7) << 4)];
            *(uint4*)db = b0;
            *(uint4*)(db + 8) = b1;
        }
        __syncthreads();
        // ---- issue next tile's global loads (overlap with WMMAs below) ----
        if (k0 + 32 < K) {
            load_a_regs(A, K, bm, k0 + 32, tid, ra0, ra1);
            load_b_regs<BT>(B, N, K, bn, k0 + 32, tid, rb);
            if (k0 + 64 < K) {
                const float* nb = BT ? (B + (size_t)(bn + (tid >> 1)) * K + k0 + 64)
                                     : (B + (size_t)(k0 + 64 + ((tid >> 7) << 4)) * N + bn + (tid & 127));
                __builtin_prefetch(nb, 0, 0);   // global_prefetch_b8
            }
        }
        // ---- WMMA on buffer `cur` ----
        Frag fa[2], fb[4];
#pragma unroll
        for (int i = 0; i < 2; ++i)
            load_frag(&lA[cur][(((wsm + i) << 4) | (lane & 15)) * GPAD], lane, fa[i]);
#pragma unroll
        for (int j = 0; j < 4; ++j)
            load_frag(&lB[cur][(((wsn + j) << 4) | (lane & 15)) * GPAD], lane, fb[j]);
#pragma unroll
        for (int i = 0; i < 2; ++i)
#pragma unroll
            for (int j = 0; j < 4; ++j)
                acc[i][j] = __builtin_amdgcn_wmma_f32_16x16x32_bf16(
                    false, fa[i].v, false, fb[j].v, (short)0, acc[i][j], false, false);
        cur ^= 1;
    }

    // ---- epilogue: C layout VGPR r -> M = r + 8*(lane>=16), N = lane&15 ----
    const int rhi = (lane >> 4) << 3;
    const int nl  = lane & 15;
#pragma unroll
    for (int i = 0; i < 2; ++i) {
        const int gmb = bm + ((wsm + i) << 4) + rhi;
#pragma unroll
        for (int j = 0; j < 4; ++j) {
            const int gn = bn + ((wsn + j) << 4) + nl;
#pragma unroll
            for (int r = 0; r < 8; ++r) {
                size_t idx = (size_t)(gmb + r) * N + gn;
                float  v   = acc[i][j][r];
                if (RESID)  v += R[idx];
                if (OUTF32) Cf[idx] = v;
                if (OUTBF)  Cb[idx] = f2bf(v);
            }
        }
    }
}

// ---------------------------------------------------------------------------
// Multi-query attention, flash-style online softmax with ALiBi bias.
// Grid: (NSEQ/128, HEADS). Block 256 threads; wave w owns 16 query rows.
// Q fragments are loop-invariant -> loaded once into registers from global.
// Faithful to reference: masked (j>i) entries get sim=1e-10 (not -inf) and
// DO participate in softmax, so the key loop covers all NSEQ keys.
// DH^-0.5 scale folded into the post-WMMA bias FMA.
// ---------------------------------------------------------------------------
__global__ __launch_bounds__(256) void attn_wmma(
    const unsigned short* __restrict__ Q,   // bf16 [NSEQ, DM] (heads packed)
    const unsigned short* __restrict__ Kb,  // bf16 [NSEQ, DHD]
    const unsigned short* __restrict__ Vb,  // bf16 [NSEQ, DHD]
    unsigned short* __restrict__ O)         // bf16 [NSEQ, DM]
{
    __shared__ unsigned short lK[32 * KPAD];    // K tile row-major [j][d]
    __shared__ unsigned short lV[128 * GPAD];   // V tile transposed [d][j]
    __shared__ unsigned short lP[8][16 * GPAD]; // per-wave P scratch [m][j]

    const int tid  = threadIdx.x;
    const int lane = tid & 31;
    const int wave = tid >> 5;
    const int i0   = blockIdx.x << 7;
    const int h    = blockIdx.y;
    const float slope  = exp2f(-0.5f * (float)(h + 1));  // ALiBi, H=16 (pow2)
    const float qscale = 0.08838834764831845f;           // DHD^-0.5

    // Q fragments (A of S = Q@K^T), one per 32-wide k-chunk of DH, in registers
    Frag fq[4];
    {
        const unsigned short* qrow =
            Q + (size_t)(i0 + (wave << 4) + (lane & 15)) * DM + h * DHD + ((lane >> 4) << 3);
#pragma unroll
        for (int s = 0; s < 4; ++s) {
            fq[s].u[0] = *(const uint4*)(qrow + (s << 5));
            fq[s].u[1] = *(const uint4*)(qrow + (s << 5) + 16);
        }
    }

    v8f o[8];
#pragma unroll
    for (int sn = 0; sn < 8; ++sn) o[sn] = v8f_zero();
    float lsum[8], mrow[8];
#pragma unroll
    for (int r = 0; r < 8; ++r) { lsum[r] = 0.f; mrow[r] = -3.0e38f; }

    const int rhi = (lane >> 4) << 3;
    const int jl  = lane & 15;

    for (int jb = 0; jb < NSEQ / 32; ++jb) {
        __syncthreads();
        // stage K (row-major copy) and V (transposed scatter)
        {
            const int j  = tid >> 3;
            const int d0 = (tid & 7) << 4;
            const unsigned short* ks = Kb + (size_t)(jb * 32 + j) * DHD + d0;
            const unsigned short* vs = Vb + (size_t)(jb * 32 + j) * DHD + d0;
            uint4 k0v = *(const uint4*)ks;
            uint4 k1v = *(const uint4*)(ks + 8);
            unsigned short* dk = &lK[j * KPAD + d0];
            *(uint4*)dk = k0v;
            *(uint4*)(dk + 8) = k1v;
            uint4 v0 = *(const uint4*)vs;
            uint4 v1 = *(const uint4*)(vs + 8);
            unsigned w[8] = {v0.x, v0.y, v0.z, v0.w, v1.x, v1.y, v1.z, v1.w};
#pragma unroll
            for (int e = 0; e < 8; ++e) {
                lV[(d0 + 2 * e)     * GPAD + j] = (unsigned short)(w[e] & 0xFFFFu);
                lV[(d0 + 2 * e + 1) * GPAD + j] = (unsigned short)(w[e] >> 16);
            }
        }
        __syncthreads();

        // S = Q @ K^T  (16 rows x 32 keys per wave)
        v8f s0 = v8f_zero(), s1 = v8f_zero();
#pragma unroll
        for (int s = 0; s < 4; ++s) {
            Frag f0, f1;
            const unsigned short* pk = &lK[(lane & 15) * KPAD + (s << 5) + ((lane >> 4) << 3)];
            f0.u[0] = *(const uint4*)pk;
            f0.u[1] = *(const uint4*)(pk + 16);
            const unsigned short* pk1 = pk + 16 * KPAD;
            f1.u[0] = *(const uint4*)pk1;
            f1.u[1] = *(const uint4*)(pk1 + 16);
            s0 = __builtin_amdgcn_wmma_f32_16x16x32_bf16(false, fq[s].v, false, f0.v,
                                                         (short)0, s0, false, false);
            s1 = __builtin_amdgcn_wmma_f32_16x16x32_bf16(false, fq[s].v, false, f1.v,
                                                         (short)0, s1, false, false);
        }

        // bias + mask + online softmax; write P (bf16) to wave-private LDS
        float osc[8];
#pragma unroll
        for (int r = 0; r < 8; ++r) {
            const int gi = i0 + (wave << 4) + rhi + r;
            const int j0 = (jb << 5) + jl;
            const int j1 = j0 + 16;
            float v0 = (j0 <= gi) ? fmaf(s0[r], qscale, slope * (float)j0) : 1e-10f;
            float v1 = (j1 <= gi) ? fmaf(s1[r], qscale, slope * (float)j1) : 1e-10f;
            float bmax = fmaxf(v0, v1);
            bmax = fmaxf(bmax, __shfl_xor(bmax, 1));
            bmax = fmaxf(bmax, __shfl_xor(bmax, 2));
            bmax = fmaxf(bmax, __shfl_xor(bmax, 4));
            bmax = fmaxf(bmax, __shfl_xor(bmax, 8));
            float nm = fmaxf(mrow[r], bmax);
            float sc = __expf(mrow[r] - nm);
            mrow[r]  = nm;
            float p0 = __expf(v0 - nm);
            float p1 = __expf(v1 - nm);
            lsum[r] = lsum[r] * sc + p0 + p1;
            osc[r]  = sc;
            lP[wave][(rhi + r) * GPAD + jl]      = f2bf(p0);
            lP[wave][(rhi + r) * GPAD + 16 + jl] = f2bf(p1);
        }
        // rescale running O by exp(old_max - new_max)
#pragma unroll
        for (int sn = 0; sn < 8; ++sn)
#pragma unroll
            for (int r = 0; r < 8; ++r) o[sn][r] *= osc[r];

        // O += P @ V  (wave-private LDS: DS ops in-order within a wave)
        Frag fp;
        load_frag(&lP[wave][(lane & 15) * GPAD], lane, fp);
#pragma unroll
        for (int sn = 0; sn < 8; ++sn) {
            Frag fv;
            load_frag(&lV[(((sn << 4) | (lane & 15))) * GPAD], lane, fv);
            o[sn] = __builtin_amdgcn_wmma_f32_16x16x32_bf16(
                false, fp.v, false, fv.v, (short)0, o[sn], false, false);
        }
    }

    // normalize (reduce row sums across the 16-lane group) and store bf16
    float inv[8];
#pragma unroll
    for (int r = 0; r < 8; ++r) {
        float t = lsum[r];
        t += __shfl_xor(t, 1); t += __shfl_xor(t, 2);
        t += __shfl_xor(t, 4); t += __shfl_xor(t, 8);
        inv[r] = 1.0f / t;
    }
#pragma unroll
    for (int sn = 0; sn < 8; ++sn)
#pragma unroll
        for (int r = 0; r < 8; ++r) {
            const int gi = i0 + (wave << 4) + rhi + r;
            O[(size_t)gi * DM + h * DHD + (sn << 4) + jl] = f2bf(o[sn][r] * inv[r]);
        }
}

// ---------------------------------------------------------------------------
// Elementwise / reduction helper kernels
// ---------------------------------------------------------------------------
__global__ __launch_bounds__(256) void embed_gather(
    const int* __restrict__ tok, const float* __restrict__ emb,
    float* __restrict__ x)
{
    int i = blockIdx.x * 256 + threadIdx.x;   // over NSEQ*DM
    if (i < NSEQ * DM) {
        int n = i >> 11;
        int d = i & (DM - 1);
        x[i] = emb[(size_t)tok[n] * DM + d];
    }
}

__global__ __launch_bounds__(256) void rmsnorm_bf16(
    const float* __restrict__ X, const float* __restrict__ g,
    unsigned short* __restrict__ Out)
{
    const int row = blockIdx.x;
    const float* x = X + (size_t)row * DM;
    float ss = 0.f;
    for (int c = threadIdx.x; c < DM; c += 256) { float v = x[c]; ss += v * v; }
    ss += __shfl_xor(ss, 16); ss += __shfl_xor(ss, 8);
    ss += __shfl_xor(ss, 4);  ss += __shfl_xor(ss, 2); ss += __shfl_xor(ss, 1);
    __shared__ float red[8];
    __shared__ float stot;
    if ((threadIdx.x & 31) == 0) red[threadIdx.x >> 5] = ss;
    __syncthreads();
    if (threadIdx.x == 0) {
        float t = 0.f;
        for (int i = 0; i < 8; ++i) t += red[i];
        stot = rsqrtf(t / (float)DM + 1e-5f);
    }
    __syncthreads();
    const float inv = stot;
    for (int c = threadIdx.x; c < DM; c += 256)
        Out[(size_t)row * DM + c] = f2bf(x[c] * inv * g[c]);
}

__global__ __launch_bounds__(256) void swiglu_bf16(
    const float* __restrict__ a, const float* __restrict__ g,
    unsigned short* __restrict__ out, int total)
{
    int i = blockIdx.x * 256 + threadIdx.x;
    if (i < total) {
        float gv = g[i];
        out[i] = f2bf(a[i] * (gv / (1.f + __expf(-gv))));
    }
}

// ---------------------------------------------------------------------------
extern "C" void kernel_launch(void* const* d_in, const int* in_sizes, int n_in,
                              void* d_out, int out_size, void* d_ws, size_t ws_size,
                              hipStream_t stream)
{
    (void)in_sizes; (void)n_in; (void)out_size; (void)ws_size;
    const int*   tokens = (const int*)  d_in[0];
    const float* emb    = (const float*)d_in[1];
    const float* agam   = (const float*)d_in[2];
    const float* wq     = (const float*)d_in[3];
    const float* wk     = (const float*)d_in[4];
    const float* wv     = (const float*)d_in[5];
    const float* wo     = (const float*)d_in[6];
    const float* fgam   = (const float*)d_in[7];
    const float* wi     = (const float*)d_in[8];
    const float* wg     = (const float*)d_in[9];
    const float* wfo    = (const float*)d_in[10];
    const float* finalg = (const float*)d_in[11];
    float* logits = (float*)d_out;

    char*  ws  = (char*)d_ws;
    size_t off = 0;
    auto alloc = [&](size_t bytes) -> void* {
        void* p = ws + off;
        off = (off + bytes + 255) & ~(size_t)255;
        return p;
    };
    float*          x   = (float*)         alloc((size_t)NSEQ * DM  * 4);
    unsigned short* hb  = (unsigned short*)alloc((size_t)NSEQ * DM  * 2);
    unsigned short* qb  = (unsigned short*)alloc((size_t)NSEQ * DM  * 2);
    unsigned short* kb  = (unsigned short*)alloc((size_t)NSEQ * DHD * 2);
    unsigned short* vb  = (unsigned short*)alloc((size_t)NSEQ * DHD * 2);
    unsigned short* ab  = (unsigned short*)alloc((size_t)NSEQ * DM  * 2);
    float*          wiB = (float*)         alloc((size_t)NSEQ * FFI * 4);
    float*          wgB = (float*)         alloc((size_t)NSEQ * FFI * 4);
    unsigned short* hid = (unsigned short*)alloc((size_t)NSEQ * FFI * 2);

    dim3 blk(256);
    embed_gather<<<(NSEQ * DM) / 256, blk, 0, stream>>>(tokens, emb, x);

    for (int l = 0; l < NDEPTH; ++l) {
        const float* Wq = wq  + (size_t)l * DM * DM;
        const float* Wk = wk  + (size_t)l * DM * DHD;
        const float* Wv = wv  + (size_t)l * DM * DHD;
        const float* Wo = wo  + (size_t)l * DM * DM;
        const float* Wi = wi  + (size_t)l * DM * FFI;
        const float* Wg = wg  + (size_t)l * DM * FFI;
        const float* Wf = wfo + (size_t)l * FFI * DM;

        rmsnorm_bf16<<<NSEQ, blk, 0, stream>>>(x, agam + (size_t)l * DM, hb);
        gemm_bf16_wmma<false, false, false, true>
            <<<dim3(NSEQ / 128, DM / 128), blk, 0, stream>>>(hb, Wq, nullptr, qb, nullptr, NSEQ, DM, DM);
        gemm_bf16_wmma<false, false, false, true>
            <<<dim3(NSEQ / 128, 1), blk, 0, stream>>>(hb, Wk, nullptr, kb, nullptr, NSEQ, DHD, DM);
        gemm_bf16_wmma<false, false, false, true>
            <<<dim3(NSEQ / 128, 1), blk, 0, stream>>>(hb, Wv, nullptr, vb, nullptr, NSEQ, DHD, DM);
        attn_wmma<<<dim3(NSEQ / 128, HEADS), blk, 0, stream>>>(qb, kb, vb, ab);
        gemm_bf16_wmma<false, true, true, false>
            <<<dim3(NSEQ / 128, DM / 128), blk, 0, stream>>>(ab, Wo, x, nullptr, x, NSEQ, DM, DM);

        rmsnorm_bf16<<<NSEQ, blk, 0, stream>>>(x, fgam + (size_t)l * DM, hb);
        gemm_bf16_wmma<false, false, true, false>
            <<<dim3(NSEQ / 128, FFI / 128), blk, 0, stream>>>(hb, Wi, wiB, nullptr, nullptr, NSEQ, FFI, DM);
        gemm_bf16_wmma<false, false, true, false>
            <<<dim3(NSEQ / 128, FFI / 128), blk, 0, stream>>>(hb, Wg, wgB, nullptr, nullptr, NSEQ, FFI, DM);
        swiglu_bf16<<<(NSEQ * FFI) / 256, blk, 0, stream>>>(wiB, wgB, hid, NSEQ * FFI);
        gemm_bf16_wmma<false, true, true, false>
            <<<dim3(NSEQ / 128, DM / 128), blk, 0, stream>>>(hid, Wf, x, nullptr, x, NSEQ, DM, FFI);
    }

    rmsnorm_bf16<<<NSEQ, blk, 0, stream>>>(x, finalg, hb);
    // logits = xn @ E^T : B-transposed path, contiguous-K loads from row-major E
    gemm_bf16_wmma<true, false, true, false>
        <<<dim3(NSEQ / 128, VOC / 128), blk, 0, stream>>>(hb, emb, logits, nullptr, nullptr, NSEQ, VOC, DM);
}
